// HierarchicalTransformerGNN_42795054137755
// MI455X (gfx1250) — compile-verified, matrix operand order checked by reading, not run on hardware
//
#include <hip/hip_runtime.h>
#include <hip/hip_bf16.h>
#include <math.h>

typedef _Float16 f16;
typedef __attribute__((ext_vector_type(16))) _Float16 v16h;
typedef __attribute__((ext_vector_type(8)))  float    v8f;

#define HEADS 8
#define LN_EPS 1e-5f
#define NEG_SLOPE 0.2f

// K position of half element j (0..15) of a 16x32 f16 A/B fragment, lane group g (0/1).
// ISA 7.12.2: VGPR0..3 -> K 0..15 (g adds +8), VGPR4..7 -> K 16..31.
__device__ __forceinline__ int afk(int j, int g) {
    return ((j >= 8) ? 16 : 0) + (((j >> 1) & 3) << 1) + (j & 1) + (g << 3);
}

// ---------------------------------------------------------------------------
// WMMA GEMM: C[M,Nout] = act(A[M,K] * B + bias)
// B element (k,n) = B[k*bsK + n*bsN]  (covers W and W^T storage)
// One wave per 32x32 output tile (2x2 fragments -> 4 WMMA per 32-K step,
// 2x reuse of both A and B fragments). K % 32 == 0, Nout % 32 == 0, M % 32 == 0.
// ---------------------------------------------------------------------------
__global__ void k_gemm(const float* __restrict__ A, long lda,
                       const float* __restrict__ B, long bsK, long bsN,
                       const float* __restrict__ bias,
                       float* __restrict__ C, long ldc,
                       int Kdim, int act) {
    const int n0 = blockIdx.x * 32;
    const int m0 = blockIdx.y * 32;
    const int lane = threadIdx.x;
    const int g = lane >> 4;
    const int q = lane & 15;

    v8f acc[2][2] = {{{}, {}}, {{}, {}}};

    for (int k0 = 0; k0 < Kdim; k0 += 32) {
        v16h a[2], b[2];
#pragma unroll
        for (int mi = 0; mi < 2; ++mi) {
            const float* Ap = A + (long)(m0 + mi * 16 + q) * lda + k0;
#pragma unroll
            for (int j = 0; j < 16; ++j) a[mi][j] = (f16)Ap[afk(j, g)];
        }
#pragma unroll
        for (int ni = 0; ni < 2; ++ni) {
            const float* Bp = B + (long)k0 * bsK + (long)(n0 + ni * 16 + q) * bsN;
#pragma unroll
            for (int j = 0; j < 16; ++j) b[ni][j] = (f16)Bp[(long)afk(j, g) * bsK];
        }
#pragma unroll
        for (int mi = 0; mi < 2; ++mi)
#pragma unroll
            for (int ni = 0; ni < 2; ++ni)
                acc[mi][ni] = __builtin_amdgcn_wmma_f32_16x16x32_f16(
                    false, a[mi], false, b[ni], (short)0, acc[mi][ni], false, false);
    }

#pragma unroll
    for (int ni = 0; ni < 2; ++ni) {
        const int col = n0 + ni * 16 + q;
        const float bv = bias ? bias[col] : 0.0f;
#pragma unroll
        for (int mi = 0; mi < 2; ++mi) {
#pragma unroll
            for (int i = 0; i < 8; ++i) {
                int row = m0 + mi * 16 + i + 8 * g;
                float v = acc[mi][ni][i] + bv;
                v = (act == 1) ? fmaxf(v, 0.0f) : v;
                C[(long)row * ldc + col] = v;
            }
        }
    }
}

// ---------------------------------------------------------------------------
// Flash attention: one wave handles 16 query rows of one head.
// qkv: [N, 3d] rows = [q | k | v]; o: [N, d]. hd <= 32, N % 32 == 0.
// ---------------------------------------------------------------------------
__global__ void k_attn(const float* __restrict__ qkv, float* __restrict__ o,
                       int n, int d, int hd, float scale) {
    const int h  = blockIdx.x;
    const int q0 = blockIdx.y * 16;
    const int lane = threadIdx.x;
    const int g = lane >> 4;
    const int q = lane & 15;
    const long ld = 3L * d;
    const long qoff = (long)h * hd;
    const long koff = (long)d + h * hd;
    const long voff = 2L * d + h * hd;
    const int nchunks = (hd + 15) / 16;

    __shared__ f16 ptile[16 * 32];

    // Q fragment (scaled), zero-padded to K=32
    v16h qa;
#pragma unroll
    for (int j = 0; j < 16; ++j) {
        int kk = afk(j, g);
        qa[j] = (kk < hd) ? (f16)(qkv[(long)(q0 + q) * ld + qoff + kk] * scale)
                          : (f16)0.0f;
    }

    v8f acc0 = {}, acc1 = {};
    float mrun[8], lrun[8];
#pragma unroll
    for (int i = 0; i < 8; ++i) { mrun[i] = -INFINITY; lrun[i] = 0.0f; }

    for (int j0 = 0; j0 < n; j0 += 32) {
        // K^T fragments for keys j0..j0+15 and j0+16..j0+31
        v16h kb0, kb1;
#pragma unroll
        for (int j = 0; j < 16; ++j) {
            int kk = afk(j, g);
            kb0[j] = (kk < hd) ? (f16)qkv[(long)(j0 + q) * ld + koff + kk] : (f16)0.0f;
            kb1[j] = (kk < hd) ? (f16)qkv[(long)(j0 + 16 + q) * ld + koff + kk] : (f16)0.0f;
        }
        v8f z = {};
        v8f s0 = __builtin_amdgcn_wmma_f32_16x16x32_f16(false, qa, false, kb0, (short)0, z, false, false);
        v8f s1 = __builtin_amdgcn_wmma_f32_16x16x32_f16(false, qa, false, kb1, (short)0, z, false, false);

#pragma unroll
        for (int i = 0; i < 8; ++i) {
            float x0 = s0[i], x1 = s1[i];
            float mx = fmaxf(x0, x1);
#pragma unroll
            for (int m = 1; m < 16; m <<= 1) mx = fmaxf(mx, __shfl_xor(mx, m, 16));
            float mnew = fmaxf(mrun[i], mx);
            float p0 = __expf(x0 - mnew), p1 = __expf(x1 - mnew);
            float corr = __expf(mrun[i] - mnew);
            float ps = p0 + p1;
#pragma unroll
            for (int m = 1; m < 16; m <<= 1) ps += __shfl_xor(ps, m, 16);
            lrun[i] = lrun[i] * corr + ps;
            mrun[i] = mnew;
            acc0[i] *= corr;
            acc1[i] *= corr;
            int row = i + 8 * g;
            ptile[row * 32 + q]      = (f16)p0;
            ptile[row * 32 + 16 + q] = (f16)p1;
        }
        __syncthreads();

        // Re-layout P (C-layout) -> A fragment via LDS
        v16h pa;
#pragma unroll
        for (int j = 0; j < 16; ++j) pa[j] = ptile[q * 32 + afk(j, g)];

        // O += P * V
        {
            v16h vb;
#pragma unroll
            for (int j = 0; j < 16; ++j) {
                int kk = afk(j, g);
                vb[j] = (q < hd) ? (f16)qkv[(long)(j0 + kk) * ld + voff + q] : (f16)0.0f;
            }
            acc0 = __builtin_amdgcn_wmma_f32_16x16x32_f16(false, pa, false, vb, (short)0, acc0, false, false);
        }
        if (nchunks > 1) {
            v16h vb;
#pragma unroll
            for (int j = 0; j < 16; ++j) {
                int kk = afk(j, g);
                int col = 16 + q;
                vb[j] = (col < hd) ? (f16)qkv[(long)(j0 + kk) * ld + voff + col] : (f16)0.0f;
            }
            acc1 = __builtin_amdgcn_wmma_f32_16x16x32_f16(false, pa, false, vb, (short)0, acc1, false, false);
        }
        __syncthreads();
    }

#pragma unroll
    for (int i = 0; i < 8; ++i) {
        int row = q0 + i + 8 * g;
        float inv = 1.0f / lrun[i];
        if (q < hd)
            o[(long)row * d + h * hd + q] = acc0[i] * inv;
        if (nchunks > 1 && 16 + q < hd)
            o[(long)row * d + h * hd + 16 + q] = acc1[i] * inv;
    }
}

// ---------------------------------------------------------------------------
// LayerNorm over last dim (optionally with residual add). blockDim.x == d.
// ---------------------------------------------------------------------------
__global__ void k_ln(const float* __restrict__ in, const float* __restrict__ res,
                     const float* __restrict__ gamma, const float* __restrict__ beta,
                     float* __restrict__ out, int d) {
    __shared__ float sd[256];
    const int row = blockIdx.x;
    const int t = threadIdx.x;
    float v = in[(long)row * d + t] + (res ? res[(long)row * d + t] : 0.0f);
    sd[t] = v; __syncthreads();
    for (int s = d >> 1; s > 0; s >>= 1) { if (t < s) sd[t] += sd[t + s]; __syncthreads(); }
    float mean = sd[0] / d; __syncthreads();
    float c = v - mean;
    sd[t] = c * c; __syncthreads();
    for (int s = d >> 1; s > 0; s >>= 1) { if (t < s) sd[t] += sd[t + s]; __syncthreads(); }
    float var = sd[0] / d;
    out[(long)row * d + t] = c * rsqrtf(var + LN_EPS) * gamma[t] + beta[t];
}

// ---------------------------------------------------------------------------
// GAT pieces
// ---------------------------------------------------------------------------
__device__ __forceinline__ unsigned fenc(float f) {
    unsigned u = __float_as_uint(f);
    return (u & 0x80000000u) ? ~u : (u | 0x80000000u);
}
__device__ __forceinline__ float fdec(unsigned u) {
    return (u & 0x80000000u) ? __uint_as_float(u & 0x7FFFFFFFu) : __uint_as_float(~u);
}
__device__ __forceinline__ void edge_sd(const int* ei, int e, int E, int& s, int& d2) {
    if (e < E) { s = ei[e]; d2 = ei[E + e]; } else { s = d2 = e - E; }
}

__global__ void k_logits(const float* __restrict__ h, const float* __restrict__ a_src,
                         const float* __restrict__ a_dst, float* __restrict__ ls,
                         float* __restrict__ ldst, int n, int hd, int d) {
    int idx = blockIdx.x * blockDim.x + threadIdx.x;
    if (idx >= n * HEADS) return;
    int nn = idx / HEADS, hh = idx % HEADS;
    float ss = 0.0f, sd = 0.0f;
    const float* hp = h + (long)nn * d + hh * hd;
    for (int k = 0; k < hd; ++k) {
        ss += hp[k] * a_src[hh * hd + k];
        sd += hp[k] * a_dst[hh * hd + k];
    }
    ls[idx] = ss; ldst[idx] = sd;
}

__global__ void k_gat_init(unsigned* __restrict__ emax, float* __restrict__ denom,
                           float* __restrict__ outraw, int n, int d) {
    int tid = blockIdx.x * blockDim.x + threadIdx.x;
    if (tid < n * d) outraw[tid] = 0.0f;
    if (tid < n * HEADS) { emax[tid] = fenc(-INFINITY); denom[tid] = 0.0f; }
}

__global__ void k_edge_max(const int* __restrict__ ei, int E, int E2,
                           const float* __restrict__ ls, const float* __restrict__ ldst,
                           unsigned* __restrict__ emax) {
    int idx = blockIdx.x * blockDim.x + threadIdx.x;
    if (idx >= E2 * HEADS) return;
    int e = idx / HEADS, hh = idx % HEADS, s, d2;
    edge_sd(ei, e, E, s, d2);
    float v = ls[s * HEADS + hh] + ldst[d2 * HEADS + hh];
    v = (v > 0.0f) ? v : NEG_SLOPE * v;
    atomicMax(&emax[d2 * HEADS + hh], fenc(v));
}

__global__ void k_edge_sum(const int* __restrict__ ei, int E, int E2,
                           const float* __restrict__ ls, const float* __restrict__ ldst,
                           const unsigned* __restrict__ emax, float* __restrict__ denom) {
    int idx = blockIdx.x * blockDim.x + threadIdx.x;
    if (idx >= E2 * HEADS) return;
    int e = idx / HEADS, hh = idx % HEADS, s, d2;
    edge_sd(ei, e, E, s, d2);
    float v = ls[s * HEADS + hh] + ldst[d2 * HEADS + hh];
    v = (v > 0.0f) ? v : NEG_SLOPE * v;
    atomicAdd(&denom[d2 * HEADS + hh], __expf(v - fdec(emax[d2 * HEADS + hh])));
}

__global__ void k_edge_scatter(const int* __restrict__ ei, int E, int E2,
                               const float* __restrict__ ls, const float* __restrict__ ldst,
                               const unsigned* __restrict__ emax,
                               const float* __restrict__ h, float* __restrict__ outraw,
                               int hd, int d) {
    int idx = blockIdx.x * blockDim.x + threadIdx.x;
    if (idx >= E2 * HEADS) return;
    int e = idx / HEADS, hh = idx % HEADS, s, d2;
    edge_sd(ei, e, E, s, d2);
    float v = ls[s * HEADS + hh] + ldst[d2 * HEADS + hh];
    v = (v > 0.0f) ? v : NEG_SLOPE * v;
    float w = __expf(v - fdec(emax[d2 * HEADS + hh]));
    const float* hp = h + (long)s * d + hh * hd;
    float* op = outraw + (long)d2 * d + hh * hd;
    for (int k = 0; k < hd; ++k) atomicAdd(&op[k], w * hp[k]);
}

__global__ void k_gat_fin(const float* __restrict__ outraw, const float* __restrict__ denom,
                          const float* __restrict__ bias, float* __restrict__ xo,
                          int n, int d, int hd) {
    int tid = blockIdx.x * blockDim.x + threadIdx.x;
    if (tid >= n * d) return;
    int nn = tid / d, c = tid % d, hh = c / hd;
    float v = outraw[tid] / denom[nn * HEADS + hh] + bias[c];
    xo[tid] = (v > 0.0f) ? v : 0.0f;
}

// ---------------------------------------------------------------------------
// Column mean + classifier head
// ---------------------------------------------------------------------------
__global__ void k_colmean(const float* __restrict__ x, float* __restrict__ out, int n, int d) {
    __shared__ float sd[256];
    int c = blockIdx.x, t = threadIdx.x;
    float s = 0.0f;
    for (int r = t; r < n; r += 256) s += x[(long)r * d + c];
    sd[t] = s; __syncthreads();
    for (int w = 128; w > 0; w >>= 1) { if (t < w) sd[t] += sd[t + w]; __syncthreads(); }
    if (t == 0) out[c] = sd[0] / n;
}

__global__ void k_classifier(const float* __restrict__ mean,
                             const float* __restrict__ Wc1, const float* __restrict__ bc1,
                             const float* __restrict__ Wc2, const float* __restrict__ bc2,
                             float* __restrict__ out) {
    __shared__ float h1[32];
    int t = threadIdx.x;
    if (t < 32) {
        float s = bc1[t];
        for (int k = 0; k < 64; ++k) s += mean[k] * Wc1[k * 32 + t];
        h1[t] = (s > 0.0f) ? s : 0.0f;
    }
    __syncthreads();
    if (t < 4) {
        float s = bc2[t];
        for (int j = 0; j < 32; ++j) s += h1[j] * Wc2[j * 4 + t];
        out[t] = s;
    }
}

// ---------------------------------------------------------------------------
// Host orchestration
// ---------------------------------------------------------------------------
static inline int ceil_div(int a, int b) { return (a + b - 1) / b; }

extern "C" void kernel_launch(void* const* d_in, const int* in_sizes, int n_in,
                              void* d_out, int out_size, void* d_ws, size_t ws_size,
                              hipStream_t stream) {
    (void)n_in; (void)out_size; (void)ws_size;
    const int IN = 768;
    const int N = in_sizes[0] / IN;       // 4096
    const int E = in_sizes[1] / 2;        // 131072
    const int E2 = E + N;

    const float* x_in = (const float*)d_in[0];
    const int* ei = (const int*)d_in[1];
    auto P = [&](int i) { return (const float*)d_in[i]; };

    // workspace layout (floats); UNIT = N*256
    const long UNIT = (long)N * 256;
    float* ws = (float*)d_ws;
    float* bx0  = ws;                 // x ping
    float* bx1  = ws + 1 * UNIT;      // x pong
    float* bh   = ws + 2 * UNIT;      // h / attn_out / ff2
    float* bqkv = ws + 3 * UNIT;      // [N, 3d]  (spans 3 units)
    float* bo   = ws + 6 * UNIT;      // attn o / GAT out_raw
    float* bff  = ws + 7 * UNIT;      // [N, 2d]  (spans 2 units)
    float* small = ws + 9 * UNIT;
    float* ls    = small;             // [N, HEADS]
    float* ldst  = small + (long)N * HEADS;
    unsigned* emax = (unsigned*)(small + 2L * N * HEADS);
    float* denom = small + 3L * N * HEADS;
    float* mean  = small + 4L * N * HEADS;

    auto gemm = [&](const float* A, long lda, const float* B, long bsK, long bsN,
                    const float* bias, float* C, long ldc, int M, int Nout, int K, int act) {
        k_gemm<<<dim3(Nout / 32, M / 32), 32, 0, stream>>>(A, lda, B, bsK, bsN, bias, C, ldc, K, act);
    };

    // ---- input projection + LN: x = LN(relu(x @ W_in + b_in)) ----
    gemm(x_in, IN, P(2), 256, 1, P(3), bx1, 256, N, 256, IN, /*relu*/1);
    k_ln<<<N, 256, 0, stream>>>(bx1, nullptr, P(4), P(5), bx0, 256);

    const int dims[3] = {256, 128, 64};
    const int base[3] = {6, 24, 42};

    for (int L = 0; L < 3; ++L) {
        const int d = dims[L];
        const int hd = d / HEADS;
        const int b = base[L];
        const int tEdge = ceil_div(E2 * HEADS, 256);
        const int tNd = ceil_div(N * d, 256);

        // ---- GAT ----
        gemm(bx0, d, P(b + 0), d, 1, nullptr, bh, d, N, d, d, 0);             // h = x @ gat_W
        k_logits<<<ceil_div(N * HEADS, 256), 256, 0, stream>>>(bh, P(b + 1), P(b + 2), ls, ldst, N, hd, d);
        k_gat_init<<<tNd, 256, 0, stream>>>(emax, denom, bo, N, d);
        k_edge_max<<<tEdge, 256, 0, stream>>>(ei, E, E2, ls, ldst, emax);
        k_edge_sum<<<tEdge, 256, 0, stream>>>(ei, E, E2, ls, ldst, emax, denom);
        k_edge_scatter<<<tEdge, 256, 0, stream>>>(ei, E, E2, ls, ldst, emax, bh, bo, hd, d);
        k_gat_fin<<<tNd, 256, 0, stream>>>(bo, denom, P(b + 3), bx1, N, d, hd);  // x_gat (relu) -> bx1

        // ---- MHA (flash) ----
        gemm(bx1, d, P(b + 4), 1, d, P(b + 5), bqkv, 3 * d, N, 3 * d, d, 0);  // qkv = x @ W^T + b
        k_attn<<<dim3(HEADS, N / 16), 32, 0, stream>>>(bqkv, bo, N, d, hd, rsqrtf((float)hd));
        gemm(bo, d, P(b + 6), 1, d, P(b + 7), bh, d, N, d, d, 0);             // out_proj
        k_ln<<<N, d, 0, stream>>>(bh, bx1, P(b + 8), P(b + 9), bx0, d);       // x = LN(x + mha)

        // ---- FFN ----
        gemm(bx0, d, P(b + 12), 1, d, P(b + 13), bff, 2 * d, N, 2 * d, d, 1); // relu(x@ff1^T+b)
        gemm(bff, 2 * d, P(b + 14), 1, 2 * d, P(b + 15), bh, d, N, d, 2 * d, 0);
        k_ln<<<N, d, 0, stream>>>(bh, bx0, P(b + 10), P(b + 11), bx1, d);     // x = LN(x + ff)

        // ---- pool ----
        if (L < 2) {
            gemm(bx1, d, P(b + 16), d / 2, 1, P(b + 17), bx0, d / 2, N, d / 2, d, 1);
        }
    }

    // ---- readout: mean over nodes, classifier ----
    k_colmean<<<64, 256, 0, stream>>>(bx1, mean, N, 64);
    k_classifier<<<1, 64, 0, stream>>>(mean, P(58), P(59), P(60), P(61), (float*)d_out);
}
